// CaC_50637664420271
// MI455X (gfx1250) — compile-verified, matrix operand order checked by reading, not run on hardware
//
#include <hip/hip_runtime.h>
#include <hip/hip_bf16.h>

#define NB    16
#define CCH   256
#define HWSZ  4096
#define NS    9
#define BN_EPS 1e-5f

typedef __attribute__((ext_vector_type(16))) __bf16 v16bf;
typedef __attribute__((ext_vector_type(8)))  float  v8f;

union BF16Frag {
    v16bf v;
    uint4 q[2];
};

// ---- CDNA5 async global->LDS helpers -------------------------------------
__device__ __forceinline__ void async_copy16(const void* g, void* l) {
    // LDS byte address = low 32 bits of the generic (shared) pointer
    unsigned lofs = (unsigned)(uintptr_t)l;
    unsigned long long ga = (unsigned long long)(uintptr_t)g;
    asm volatile("global_load_async_to_lds_b128 %0, %1, off"
                 :: "v"(lofs), "v"(ga) : "memory");
}

__device__ __forceinline__ void async_wait0() {
#if __has_builtin(__builtin_amdgcn_s_wait_asynccnt)
    __builtin_amdgcn_s_wait_asynccnt(0);
#else
    asm volatile("s_wait_asynccnt 0x0" ::: "memory");
#endif
}

// ---------------------------------------------------------------------------
// K0: convert x (f32) -> bf16 rows, and compute per-row sums s[n][c]
// grid: 4096 blocks (one per (n,c) row), 256 threads
// ---------------------------------------------------------------------------
__global__ void k0_convert_rowsum(const float* __restrict__ x,
                                  __hip_bfloat16* __restrict__ xbf,
                                  float* __restrict__ rowsum) {
    const int row = blockIdx.x;          // n*CCH + c
    const int tid = threadIdx.x;
    const float4* src4 = reinterpret_cast<const float4*>(x + (size_t)row * HWSZ);
    __hip_bfloat16* dst = xbf + (size_t)row * HWSZ;

    float acc = 0.f;
    #pragma unroll
    for (int j = 0; j < 4; ++j) {
        int i4 = tid + j * 256;          // float4 index 0..1023
        float4 f = src4[i4];
        acc += f.x + f.y + f.z + f.w;
        int base = i4 * 4;
        dst[base + 0] = __float2bfloat16(f.x);
        dst[base + 1] = __float2bfloat16(f.y);
        dst[base + 2] = __float2bfloat16(f.z);
        dst[base + 3] = __float2bfloat16(f.w);
    }
    __shared__ float red[256];
    red[tid] = acc;
    __syncthreads();
    for (int s = 128; s > 0; s >>= 1) {
        if (tid < s) red[tid] += red[tid + s];
        __syncthreads();
    }
    if (tid == 0) rowsum[row] = red[0];
}

// ---------------------------------------------------------------------------
// K1: Gram matrix G_n = X_n * X_n^T.
// Block tile 64x64 (8 waves, each wave a 16x32 sub-tile, 2 accumulators).
// K consumed in 64-wide blocks; A/B panels staged in LDS with
// global_load_async_to_lds_b128 + double buffering; WMMA fed by ds_load_b128.
// LDS: 2 buffers x 2 panels x 64 rows x 72 bf16 (144B padded rows) = 36 KB.
// ---------------------------------------------------------------------------
__device__ __forceinline__ void k1_compute_block(
        const __hip_bfloat16 (*buf)[64][72],   // [panel][row][col]
        int iT, int jT, int m, int h, v8f& acc0, v8f& acc1) {
    #pragma unroll
    for (int ks = 0; ks < 2; ++ks) {
        const int kk = ks * 32;
        BF16Frag a, b0, b1;
        const __hip_bfloat16* pa = &buf[0][iT * 16 + m][kk + h * 8];
        a.q[0] = *reinterpret_cast<const uint4*>(pa);
        a.q[1] = *reinterpret_cast<const uint4*>(pa + 16);
        const __hip_bfloat16* pb0 = &buf[1][jT * 32 + m][kk + h * 8];
        b0.q[0] = *reinterpret_cast<const uint4*>(pb0);
        b0.q[1] = *reinterpret_cast<const uint4*>(pb0 + 16);
        const __hip_bfloat16* pb1 = &buf[1][jT * 32 + 16 + m][kk + h * 8];
        b1.q[0] = *reinterpret_cast<const uint4*>(pb1);
        b1.q[1] = *reinterpret_cast<const uint4*>(pb1 + 16);
        acc0 = __builtin_amdgcn_wmma_f32_16x16x32_bf16(
                   false, a.v, false, b0.v, (short)0, acc0, false, false);
        acc1 = __builtin_amdgcn_wmma_f32_16x16x32_bf16(
                   false, a.v, false, b1.v, (short)0, acc1, false, false);
    }
}

__global__ void k1_gram_wmma(const __hip_bfloat16* __restrict__ xbf,
                             float* __restrict__ G) {
    __shared__ __hip_bfloat16 lds[2][2][64][72];

    const int tid  = threadIdx.x;
    const int wave = tid >> 5;
    const int lane = tid & 31;
    const int m = lane & 15;
    const int h = lane >> 4;

    const int b  = blockIdx.x;           // 0..255
    const int n  = b >> 4;               // sample
    const int i0 = ((b >> 2) & 3) << 6;  // block row base
    const int j0 = (b & 3) << 6;         // block col base

    const int iT = wave >> 1;            // 0..3 : 16-row tile in block
    const int jT = wave & 1;             // 0..1 : 32-col group in block

    const size_t rowBaseA = (size_t)(n * CCH + i0) * HWSZ;
    const size_t rowBaseB = (size_t)(n * CCH + j0) * HWSZ;

    // this thread's 4 16-byte chunks per K-block (1024 chunks total)
    int pnl[4], crow[4], cc[4];
    #pragma unroll
    for (int t = 0; t < 4; ++t) {
        int chunk = tid + t * 256;       // 0..1023
        pnl[t]  = chunk >> 9;            // 0 = A panel, 1 = B panel
        crow[t] = (chunk >> 3) & 63;     // row in panel
        cc[t]   = (chunk & 7) * 8;       // bf16 column offset
    }

    v8f acc0 = {}, acc1 = {};

    // prologue: fill buffer 0 with K-block 0
    #pragma unroll
    for (int t = 0; t < 4; ++t) {
        const __hip_bfloat16* g = xbf + (pnl[t] ? rowBaseB : rowBaseA)
                                      + (size_t)crow[t] * HWSZ + cc[t];
        async_copy16(g, &lds[0][pnl[t]][crow[t]][cc[t]]);
    }
    async_wait0();
    __syncthreads();

    const int NKB = HWSZ / 64;           // 64 K-blocks
    #pragma unroll 1
    for (int kb = 0; kb < NKB - 1; ++kb) {
        const int cur = kb & 1;
        const int nxt = cur ^ 1;
        const int k0  = (kb + 1) * 64;
        #pragma unroll
        for (int t = 0; t < 4; ++t) {
            const __hip_bfloat16* g = xbf + (pnl[t] ? rowBaseB : rowBaseA)
                                          + (size_t)crow[t] * HWSZ + k0 + cc[t];
            async_copy16(g, &lds[nxt][pnl[t]][crow[t]][cc[t]]);
        }
        k1_compute_block(lds[cur], iT, jT, m, h, acc0, acc1);
        async_wait0();
        __syncthreads();
    }
    // final K-block (no prefetch)
    k1_compute_block(lds[(NKB - 1) & 1], iT, jT, m, h, acc0, acc1);

    // C/D layout: VGPR r, lanes 0-15 -> (M=r, N=lane); lanes 16-31 -> (M=8+r)
    float* Gn = G + (size_t)n * CCH * CCH;
    const int gr  = i0 + iT * 16 + 8 * h;
    const int gc0 = j0 + jT * 32 + m;
    #pragma unroll
    for (int r = 0; r < 8; ++r) {
        Gn[(gr + r) * CCH + gc0]      = acc0[r];
        Gn[(gr + r) * CCH + gc0 + 16] = acc1[r];
    }
}

// ---------------------------------------------------------------------------
// K2a: v[n][s][i] = sum_j wq[s][j] * G_n[j][i];  wqs[n][s] = wq[s] . s_n
// grid: 144 blocks (n,s), 256 threads (i)
// ---------------------------------------------------------------------------
__global__ void k2a_wq_gram(const float* __restrict__ G,
                            const float* __restrict__ wq,
                            const float* __restrict__ rowsum,
                            float* __restrict__ v,
                            float* __restrict__ wqs) {
    const int b   = blockIdx.x;          // n*NS + s
    const int n   = b / NS;
    const int s   = b % NS;
    const int tid = threadIdx.x;

    __shared__ float wql[256];
    __shared__ float red[256];
    float w = wq[s * CCH + tid];
    wql[tid] = w;
    red[tid] = w * rowsum[n * CCH + tid];
    __syncthreads();

    const float* Gn = G + (size_t)n * CCH * CCH;
    float acc = 0.f;
    for (int j = 0; j < CCH; ++j) acc += wql[j] * Gn[j * CCH + tid];
    v[(size_t)b * CCH + tid] = acc;

    for (int st = 128; st > 0; st >>= 1) {
        if (tid < st) red[tid] += red[tid + st];
        __syncthreads();
    }
    if (tid == 0) wqs[b] = red[0];
}

// ---------------------------------------------------------------------------
// K2b: krnl[n][c][s] = sum_i wk[c][i]*(v[n][s][i] + bq[s]*s_n[i])
//                      + bk[c]*(wqs[n][s] + hw*bq[s])
// grid: 144 blocks (n,s), 256 threads (c)
// ---------------------------------------------------------------------------
__global__ void k2b_make_kernels(const float* __restrict__ v,
                                 const float* __restrict__ rowsum,
                                 const float* __restrict__ wk,
                                 const float* __restrict__ bk,
                                 const float* __restrict__ bq,
                                 const float* __restrict__ wqs,
                                 float* __restrict__ krnl) {
    const int b = blockIdx.x;            // n*NS + s
    const int n = b / NS;
    const int s = b % NS;
    const int c = threadIdx.x;

    __shared__ float vl[256];
    const float bqs = bq[s];
    vl[c] = v[(size_t)b * CCH + c] + bqs * rowsum[n * CCH + c];
    __syncthreads();

    const float* wkr = wk + (size_t)c * CCH;
    float acc = 0.f;
    for (int i = 0; i < CCH; ++i) acc += wkr[i] * vl[i];
    acc += bk[c] * (wqs[b] + (float)HWSZ * bqs);
    krnl[(size_t)(n * CCH + c) * NS + s] = acc;
}

// ---------------------------------------------------------------------------
// K3: BatchNorm stats over (n, 3, 3) = 144 values per channel -> scale/shift
// ---------------------------------------------------------------------------
__global__ void k3_bn_stats(const float* __restrict__ krnl,
                            const float* __restrict__ gamma,
                            const float* __restrict__ beta,
                            float* __restrict__ scale,
                            float* __restrict__ shift) {
    const int c = threadIdx.x;
    float sum = 0.f, sumsq = 0.f;
    for (int n = 0; n < NB; ++n) {
        const float* p = krnl + (size_t)(n * CCH + c) * NS;
        #pragma unroll
        for (int s = 0; s < NS; ++s) {
            float vv = p[s];
            sum += vv;
            sumsq += vv * vv;
        }
    }
    const float inv_cnt = 1.f / (float)(NB * NS);
    float mean = sum * inv_cnt;
    float var  = sumsq * inv_cnt - mean * mean;
    float sc   = rsqrtf(var + BN_EPS) * gamma[c];
    scale[c] = sc;
    shift[c] = beta[c] - mean * sc;
}

// ---------------------------------------------------------------------------
// K4: depthwise dilated 3x3 convs (d=1,2,3) + sigmoid + average.
// grid: 4096 blocks (one (n,c) plane), 256 threads, plane cached in LDS (16KB)
// ---------------------------------------------------------------------------
__global__ void k4_dwconv(const float* __restrict__ x,
                          const float* __restrict__ krnl,
                          const float* __restrict__ scale,
                          const float* __restrict__ shift,
                          float* __restrict__ out) {
    const int plane = blockIdx.x;        // n*CCH + c
    const int c     = plane & (CCH - 1);
    const int tid   = threadIdx.x;

    __shared__ float p[HWSZ];
    __shared__ float wloc[NS];

    const float4* src4 = reinterpret_cast<const float4*>(x + (size_t)plane * HWSZ);
    float4* p4 = reinterpret_cast<float4*>(p);
    #pragma unroll
    for (int j = 0; j < 4; ++j) p4[tid + j * 256] = src4[tid + j * 256];

    if (tid < NS) {
        wloc[tid] = krnl[(size_t)plane * NS + tid] * scale[c] + shift[c];
    }
    __syncthreads();

    float* o = out + (size_t)plane * HWSZ;
    for (int k = 0; k < 16; ++k) {
        int pix = tid + k * 256;
        int y  = pix >> 6;
        int xx = pix & 63;
        float r = 0.f;
        #pragma unroll
        for (int d = 1; d <= 3; ++d) {
            float acc = 0.f;
            #pragma unroll
            for (int dy = -1; dy <= 1; ++dy) {
                int yy = y + dy * d;
                #pragma unroll
                for (int dx = -1; dx <= 1; ++dx) {
                    int xc = xx + dx * d;
                    float val = (yy >= 0 && yy < 64 && xc >= 0 && xc < 64)
                                    ? p[yy * 64 + xc] : 0.f;
                    acc += val * wloc[(dy + 1) * 3 + (dx + 1)];
                }
            }
            r += 1.f / (1.f + __expf(-acc));
        }
        o[pix] = r * (1.f / 3.f);
    }
}

// ---------------------------------------------------------------------------
// Workspace layout (bytes): xbf 33.5MB | G 4MB | rsum 16KB | v 144KB |
// wqs 1KB | krnl 144KB | scale 1KB | shift 1KB  (~36.3 MB total)
// ---------------------------------------------------------------------------
extern "C" void kernel_launch(void* const* d_in, const int* in_sizes, int n_in,
                              void* d_out, int out_size, void* d_ws, size_t ws_size,
                              hipStream_t stream) {
    (void)in_sizes; (void)n_in; (void)out_size; (void)ws_size;

    const float* x     = (const float*)d_in[0];
    const float* wk    = (const float*)d_in[1];
    const float* bk    = (const float*)d_in[2];
    const float* wq    = (const float*)d_in[3];
    const float* bq    = (const float*)d_in[4];
    const float* gamma = (const float*)d_in[5];
    const float* beta  = (const float*)d_in[6];
    float* out = (float*)d_out;

    char* ws = (char*)d_ws;
    __hip_bfloat16* xbf = (__hip_bfloat16*)ws;
    float* G     = (float*)(ws + 33554432);
    float* rsum  = (float*)(ws + 37748736);
    float* v     = (float*)(ws + 37765120);
    float* wqs   = (float*)(ws + 37912576);
    float* krnl  = (float*)(ws + 37913600);
    float* scale = (float*)(ws + 38061056);
    float* shift = (float*)(ws + 38062080);

    k0_convert_rowsum<<<NB * CCH, 256, 0, stream>>>(x, xbf, rsum);
    k1_gram_wmma<<<256, 256, 0, stream>>>(xbf, G);
    k2a_wq_gram<<<NB * NS, 256, 0, stream>>>(G, wq, rsum, v, wqs);
    k2b_make_kernels<<<NB * NS, 256, 0, stream>>>(v, rsum, wk, bk, bq, wqs, krnl);
    k3_bn_stats<<<1, 256, 0, stream>>>(krnl, gamma, beta, scale, shift);
    k4_dwconv<<<NB * CCH, 256, 0, stream>>>(x, krnl, scale, shift, out);
}
